// SolarSeq2SeqGRU_18365280157965
// MI455X (gfx1250) — matrix operand order repaired and checked by reading, not compiled, not run on hardware
//
#include <hip/hip_runtime.h>

#define T_PRIOR_  168
#define N_FEAT_   11
#define DEC_FEAT_ 7
#define HID_      64
#define T_DEC_    48
#define G_        192   // 3*HID

typedef __attribute__((ext_vector_type(16))) _Float16 v16h;
typedef __attribute__((ext_vector_type(8)))  float    v8f;

union Frag {
  v16h h;
  uint4 u4[2];
  unsigned int u[8];
  _Float16 f16[16];
};

__device__ __forceinline__ float fast_rcp(float x) { return __builtin_amdgcn_rcpf(x); }
__device__ __forceinline__ float sigf(float x) {
  return fast_rcp(1.0f + __expf(-x));
}
__device__ __forceinline__ float tanh_fast(float x) {
  return 2.0f * fast_rcp(1.0f + __expf(-2.0f * x)) - 1.0f;
}
__device__ __forceinline__ unsigned int pk2(float a, float b) {
  union { decltype(__builtin_amdgcn_cvt_pkrtz(0.f, 0.f)) v; unsigned int u; } t;
  t.v = __builtin_amdgcn_cvt_pkrtz(a, b);
  return t.u;
}

// Split workgroup barrier: signal as soon as the h-fragment store is visible,
// wait only after the pre-barrier WMMA burst. Mnemonics match compiler output.
__device__ __forceinline__ void pair_barrier_signal() {
  asm volatile("s_wait_dscnt 0x0\n\ts_barrier_signal -1" ::: "memory");
}
__device__ __forceinline__ void pair_barrier_wait() {
  asm volatile("s_barrier_wait -1" ::: "memory");
}

// A-fragment (16x32 f16) load from LDS row-major weights [G][KP].
__device__ __forceinline__ v16h load_a(const _Float16* w, int KP, int gt, int kt, int lane) {
  int row = gt * 16 + (lane & 15);
  int col = kt * 32 + ((lane & 16) ? 16 : 0);
  const _Float16* p = w + row * KP + col;
  Frag f;
  f.u4[0] = *(const uint4*)(p);
  f.u4[1] = *(const uint4*)(p + 8);
  return f.h;
}

// K-column permutation for Whh staging: lane-local h^T B-fragment build.
__device__ __forceinline__ int permc(int c) {
  int cc = c & 31;
  if (cc >= 8 && cc < 16) return c + 8;
  if (cc >= 16 && cc < 24) return c - 8;
  return c;
}

// Pack this wave's 2 h-tiles (C/D layout) -> B-fragment bh[half].
__device__ __forceinline__ v16h build_bh_own(const v8f hC[2]) {
  Frag f;
#pragma unroll
  for (int d = 0; d < 4; ++d) {
    f.u[d]     = pk2(hC[0][2 * d], hC[0][2 * d + 1]);
    f.u[4 + d] = pk2(hC[1][2 * d], hC[1][2 * d + 1]);
  }
  return f.h;
}

#define WMMA(c, a, b) \
  (c) = __builtin_amdgcn_wmma_f32_16x16x32_f16(false, (a), false, (b), (short)0, (c), false, false)

// All 18 weight fragments this wave ever needs, register-resident (144 VGPRs).
struct WReg {
  v16h own[2][3];   // Whh, own K-half,    [tt][r,z,n]
  v16h par[2][3];   // Whh, partner K-half [tt][r,z,n]
  v16h wx[2][3];    // Wih(+bias col),     [tt][r,z,n]
};

__device__ __forceinline__ void load_wreg(WReg& w, const _Float16* sWA,
                                          const _Float16* sWhh, int half, int lane) {
#pragma unroll
  for (int tt = 0; tt < 2; ++tt) {
    const int t = 2 * half + tt;
    w.own[tt][0] = load_a(sWhh, HID_, t,     half,     lane);
    w.own[tt][1] = load_a(sWhh, HID_, 4 + t, half,     lane);
    w.own[tt][2] = load_a(sWhh, HID_, 8 + t, half,     lane);
    w.par[tt][0] = load_a(sWhh, HID_, t,     1 - half, lane);
    w.par[tt][1] = load_a(sWhh, HID_, 4 + t, 1 - half, lane);
    w.par[tt][2] = load_a(sWhh, HID_, 8 + t, 1 - half, lane);
    w.wx[tt][0]  = load_a(sWA, 32, t,     0, lane);
    w.wx[tt][1]  = load_a(sWA, 32, 4 + t, 0, lane);
    w.wx[tt][2]  = load_a(sWA, 32, 8 + t, 0, lane);
  }
}

// x-side partial gate pre-activations (6 WMMAs, off the h critical path).
struct XAcc { v8f r[2]; v8f z[2]; v8f n[2]; };

__device__ __forceinline__ void x_partial(const WReg& w, XAcc& xa, v16h bx) {
  const v8f zf = {0.f,0.f,0.f,0.f,0.f,0.f,0.f,0.f};
#pragma unroll
  for (int tt = 0; tt < 2; ++tt) {
    xa.r[tt] = zf; xa.z[tt] = zf; xa.n[tt] = zf;
    WMMA(xa.r[tt], w.wx[tt][0], bx);
    WMMA(xa.z[tt], w.wx[tt][1], bx);
    WMMA(xa.n[tt], w.wx[tt][2], bx);   // i_n: no Whh term, this IS cin
  }
}

// Own-K h WMMAs, seeded with the x partials (between barrier signal and wait).
__device__ __forceinline__ void h_own(const WReg& w, const XAcc& xa,
                                      v8f cr[2], v8f cz[2], v8f chn[2], v16h bhOwn) {
  const v8f zf = {0.f,0.f,0.f,0.f,0.f,0.f,0.f,0.f};
#pragma unroll
  for (int tt = 0; tt < 2; ++tt) {
    cr[tt] = xa.r[tt]; cz[tt] = xa.z[tt]; chn[tt] = zf;
    WMMA(cr[tt],  w.own[tt][0], bhOwn);
    WMMA(cz[tt],  w.own[tt][1], bhOwn);
    WMMA(chn[tt], w.own[tt][2], bhOwn);
  }
}

// Partner-K h WMMAs (after the barrier wait) + gating.
__device__ __forceinline__ void h_partner_gate(const WReg& w, v8f cr[2], v8f cz[2],
                                               v8f chn[2], const v8f cin[2], v16h bhP,
                                               v8f hC[2], const float bhn[2][8]) {
#pragma unroll
  for (int tt = 0; tt < 2; ++tt) {
    WMMA(cr[tt],  w.par[tt][0], bhP);
    WMMA(cz[tt],  w.par[tt][1], bhP);
    WMMA(chn[tt], w.par[tt][2], bhP);
  }
#pragma unroll
  for (int tt = 0; tt < 2; ++tt) {
#pragma unroll
    for (int v = 0; v < 8; ++v) {
      float r = sigf(cr[tt][v]);
      float z = sigf(cz[tt][v]);
      float n = tanh_fast(cin[tt][v] + r * (chn[tt][v] + bhn[tt][v]));
      hC[tt][v] = (1.0f - z) * n + z * hC[tt][v];
    }
  }
}

__device__ __forceinline__ void stage_weights(_Float16* sWA, _Float16* sWhh,
                                              const float* Wih, int nfeat, const float* Whh,
                                              const float* bih, const float* bhh,
                                              int tid, int nthr) {
  for (int i = tid; i < G_ * 32; i += nthr) {
    int g = i >> 5, k = i & 31;
    float v = 0.0f;
    if (k < nfeat)       v = Wih[g * nfeat + k];
    else if (k == nfeat) v = (g < 2 * HID_) ? (bih[g] + bhh[g]) : bih[g];
    sWA[i] = (_Float16)v;
  }
  for (int i = tid; i < G_ * HID_; i += nthr) {
    int g = i >> 6, c = i & 63;
    sWhh[i] = (_Float16)Whh[g * HID_ + permc(c)];
  }
}

__global__ __launch_bounds__(128)
void solar_gru_kernel(const float* __restrict__ prior, const float* __restrict__ teacher,
                      const float* __restrict__ eWih, const float* __restrict__ eWhh,
                      const float* __restrict__ ebih, const float* __restrict__ ebhh,
                      const float* __restrict__ dWih, const float* __restrict__ dWhh,
                      const float* __restrict__ dbih, const float* __restrict__ dbhh,
                      const float* __restrict__ fcW, const float* __restrict__ fcb,
                      float* __restrict__ out) {
  __shared__ __align__(16) _Float16 sWA[G_ * 32];      // 12 KB
  __shared__ __align__(16) _Float16 sWhh[G_ * HID_];   // 24 KB (reused enc->dec)
  __shared__ __align__(16) uint4 sBH[2][2][2][32][2];  // [parity][tile][half][lane] 8 KB
  __shared__ float sFc[2][2][2][16];                   // [parity][tile][half][n]

  const int tid  = threadIdx.x;
  const int lane = tid & 31;
  const int wave = tid >> 5;
  const int half = wave & 1;
  const int tile = wave >> 1;
  const bool loLane = (lane & 16) == 0;
  const int off = loLane ? 0 : 8;
  const int row = blockIdx.x * 32 + tile * 16 + (lane & 15);

  // loop-invariant row base pointers (kills in-loop 64-bit muls)
  const float* __restrict__ xbase = prior + (size_t)row * T_PRIOR_ * N_FEAT_;
  const float* __restrict__ tbase = teacher + (size_t)row * T_DEC_ * DEC_FEAT_;

  // ---- stage encoder weights ----
  stage_weights(sWA, sWhh, eWih, N_FEAT_, eWhh, ebih, ebhh, tid, 128);

  float bhn[2][8];
#pragma unroll
  for (int tt = 0; tt < 2; ++tt)
#pragma unroll
    for (int v = 0; v < 8; ++v)
      bhn[tt][v] = ebhh[2 * HID_ + 16 * (2 * half + tt) + v + off];

  __syncthreads();

  WReg w;
  load_wreg(w, sWA, sWhh, half, lane);   // 18 fragments -> registers, loop-invariant

  const v8f zf = {0.f,0.f,0.f,0.f,0.f,0.f,0.f,0.f};
  v8f hC[2];
  hC[0] = zf; hC[1] = zf;

  // ---------------- encoder: 168 steps ----------------
  // x pipeline depth 2: xA = x(s+1) values, xB = x(s+2) in flight.
  // Loads are unconditional (all lanes, same cachelines) with clamped time
  // index -> always in-bounds; over-read values are provably never consumed.
  float x0[N_FEAT_], xA[N_FEAT_], xB[N_FEAT_];
#pragma unroll
  for (int k = 0; k < N_FEAT_; ++k) x0[k] = xbase[k];
#pragma unroll
  for (int k = 0; k < N_FEAT_; ++k) xA[k] = xbase[N_FEAT_ + k];

  XAcc xacc;
  {
    Frag bx;
#pragma unroll
    for (int d = 0; d < 8; ++d) bx.u[d] = 0u;
    if (loLane) {
#pragma unroll
      for (int k = 0; k < N_FEAT_; ++k) bx.f16[k] = (_Float16)x0[k];
      bx.f16[N_FEAT_] = (_Float16)1.0f;
    }
    x_partial(w, xacc, bx.h);            // x gates for step 0
  }

  for (int s = 0; s < T_PRIOR_; ++s) {
    const int par = s & 1;

    // publish own h fragment, signal the barrier immediately
    Frag bo; bo.h = build_bh_own(hC);
    sBH[par][tile][half][lane][0] = bo.u4[0];
    sBH[par][tile][half][lane][1] = bo.u4[1];
    pair_barrier_signal();

    v8f cr[2], cz[2], chn[2];
    h_own(w, xacc, cr, cz, chn, bo.h);   // 6 WMMAs, seeded with x partials

    // issue x(s+2) global loads (clamped index, always in-bounds)
    {
      const int t2 = (s + 2 < T_PRIOR_) ? (s + 2) : (T_PRIOR_ - 1);
      const float* xp = xbase + t2 * N_FEAT_;
#pragma unroll
      for (int k = 0; k < N_FEAT_; ++k) xB[k] = xp[k];
    }

    // next step's x gates (6 WMMAs, off the h critical path)
    Frag bxN;
#pragma unroll
    for (int d = 0; d < 8; ++d) bxN.u[d] = 0u;
    if (loLane) {
#pragma unroll
      for (int k = 0; k < N_FEAT_; ++k) bxN.f16[k] = (_Float16)xA[k];
      bxN.f16[N_FEAT_] = (_Float16)1.0f;
    }
    XAcc xaccN;
    x_partial(w, xaccN, bxN.h);

    pair_barrier_wait();

    Frag bp;
    bp.u4[0] = sBH[par][tile][1 - half][lane][0];
    bp.u4[1] = sBH[par][tile][1 - half][lane][1];
    h_partner_gate(w, cr, cz, chn, xacc.n, bp.h, hC, bhn);

    xacc = xaccN;
#pragma unroll
    for (int k = 0; k < N_FEAT_; ++k) xA[k] = xB[k];
  }

  // ---- re-stage LDS with decoder weights ----
  __syncthreads();
  stage_weights(sWA, sWhh, dWih, DEC_FEAT_, dWhh, dbih, dbhh, tid, 128);

#pragma unroll
  for (int tt = 0; tt < 2; ++tt)
#pragma unroll
    for (int v = 0; v < 8; ++v)
      bhn[tt][v] = dbhh[2 * HID_ + 16 * (2 * half + tt) + v + off];

  float fcwv[2][8];
#pragma unroll
  for (int tt = 0; tt < 2; ++tt)
#pragma unroll
    for (int v = 0; v < 8; ++v)
      fcwv[tt][v] = fcW[16 * (2 * half + tt) + v + off];
  const float fcB = fcb[0];

  __syncthreads();

  load_wreg(w, sWA, sWhh, half, lane);   // decoder fragments -> registers

  // ---------------- decoder: 48 autoregressive steps ----------------
  float xf[DEC_FEAT_];
#pragma unroll
  for (int k = 0; k < DEC_FEAT_; ++k) xf[k] = tbase[k];   // x0 = teacher[:,0,:]

  for (int td = 0; td < T_DEC_; ++td) {
    const int par = td & 1;

    // publish own h fragment first, signal immediately
    Frag bo; bo.h = build_bh_own(hC);
    sBH[par][tile][half][lane][0] = bo.u4[0];
    sBH[par][tile][half][lane][1] = bo.u4[1];
    pair_barrier_signal();

    // x gates in-step (pred feedback prevents pipelining here)
    Frag bx;
#pragma unroll
    for (int d = 0; d < 8; ++d) bx.u[d] = 0u;
    if (loLane) {
#pragma unroll
      for (int k = 0; k < DEC_FEAT_; ++k) bx.f16[k] = (_Float16)xf[k];
      bx.f16[DEC_FEAT_] = (_Float16)1.0f;
    }
    XAcc xa;
    x_partial(w, xa, bx.h);

    v8f cr[2], cz[2], chn[2];
    h_own(w, xa, cr, cz, chn, bo.h);

    // prefetch teacher row for td+1 (clamped; td=47 value never consumed)
    const int tn_t = (td + 1 < T_DEC_) ? (td + 1) : (T_DEC_ - 1);
    const float* tp = tbase + tn_t * DEC_FEAT_;
    const float tn0 = tp[0], tn1 = tp[1], tn2 = tp[2], tn3 = tp[3], tn5 = tp[5], tn6 = tp[6];

    pair_barrier_wait();

    Frag bp;
    bp.u4[0] = sBH[par][tile][1 - half][lane][0];
    bp.u4[1] = sBH[par][tile][1 - half][lane][1];
    h_partner_gate(w, cr, cz, chn, xa.n, bp.h, hC, bhn);

    // fc over this wave's half of the hidden units, pair-summed in-wave
    float part = 0.0f;
#pragma unroll
    for (int tt = 0; tt < 2; ++tt)
#pragma unroll
      for (int v = 0; v < 8; ++v) part += hC[tt][v] * fcwv[tt][v];
    part += __shfl_xor(part, 16, 32);
    if (loLane) sFc[par][tile][half][lane] = part;

    __syncthreads();

    float pred = part + sFc[par][tile][1 - half][lane & 15] + fcB;
    if (half == 0 && loLane) out[(size_t)row * T_DEC_ + td] = pred;

    xf[0] = tn0; xf[1] = tn1; xf[2] = tn2; xf[3] = tn3;
    xf[4] = pred; xf[5] = tn5; xf[6] = tn6;
  }
}

extern "C" void kernel_launch(void* const* d_in, const int* in_sizes, int n_in,
                              void* d_out, int out_size, void* d_ws, size_t ws_size,
                              hipStream_t stream) {
  const float* prior = (const float*)d_in[0];
  const float* teach = (const float*)d_in[1];
  const float* eWih  = (const float*)d_in[2];
  const float* eWhh  = (const float*)d_in[3];
  const float* ebih  = (const float*)d_in[4];
  const float* ebhh  = (const float*)d_in[5];
  const float* dWih  = (const float*)d_in[6];
  const float* dWhh  = (const float*)d_in[7];
  const float* dbih  = (const float*)d_in[8];
  const float* dbhh  = (const float*)d_in[9];
  const float* fcW   = (const float*)d_in[10];
  const float* fcb   = (const float*)d_in[11];
  float* out = (float*)d_out;

  const int B = in_sizes[0] / (T_PRIOR_ * N_FEAT_);   // 8192
  const int grid = B / 32;  // 2 tiles/block, 2 waves/tile, 16 rows/tile
  solar_gru_kernel<<<grid, 128, 0, stream>>>(prior, teach, eWih, eWhh, ebih, ebhh,
                                             dWih, dWhh, dbih, dbhh, fcW, fcb, out);
}
